// DeepGNNSigmoid_1872605741179
// MI455X (gfx1250) — matrix operand order repaired and checked by reading, not compiled, not run on hardware
//
#include <hip/hip_runtime.h>
#include <hip/hip_bf16.h>
#include <math.h>

// ---------------------------------------------------------------------------
// Types for CDNA5 WMMA / TDM (wave32)
// ---------------------------------------------------------------------------
typedef __attribute__((ext_vector_type(16))) __bf16 v16bf;
typedef __attribute__((ext_vector_type(8)))  __bf16 v8bf;
typedef __attribute__((ext_vector_type(8)))  float  v8f;
typedef __attribute__((ext_vector_type(4)))  unsigned int v4u;
typedef __attribute__((ext_vector_type(8)))  int v8i_t;
typedef __attribute__((ext_vector_type(4)))  int v4i_t;

static __device__ __forceinline__ unsigned short f32_to_bf16_rne(float f) {
    unsigned int u = __builtin_bit_cast(unsigned int, f);
    unsigned int r = u + 0x7FFFu + ((u >> 16) & 1u);
    return (unsigned short)(r >> 16);
}

static __device__ __forceinline__ v16bf cat8(v8bf lo, v8bf hi) {
    return __builtin_shufflevector(lo, hi, 0,1,2,3,4,5,6,7,8,9,10,11,12,13,14,15);
}

// ---------------------------------------------------------------------------
// TDM: issue a 2-D tensor_load_to_lds (bf16 elements, data_size=2B).
//   - tile (tile_d0 x tile_d1) starting at gptr, row stride stride0 elems
//   - tensor_d0/d1 = remaining extent from tile origin (OOB reads -> 0)
//   - LDS row padding via pad_interval/pad_amount codes
// D# packing per CDNA5 ISA 8.3/8.4 (group0: count=1, type=2).
// Toolchain here uses the 6-arg builtin: (g0, g1, g2, g3, g2b, cpol).
// ---------------------------------------------------------------------------
static __device__ __forceinline__ void tdm_load_2d_bf16(
    unsigned lds_off, const unsigned short* gptr,
    unsigned tensor_d0, unsigned tensor_d1,
    unsigned tile_d0, unsigned tile_d1,
    unsigned long long stride0,
    unsigned pad_interval_code, unsigned pad_amount_code)
{
    unsigned long long ga = (unsigned long long)(size_t)gptr;
    v4u g0;
    g0.x = 1u;                                   // count=1, user mode
    g0.y = lds_off;                              // LDS byte address
    g0.z = (unsigned)ga;                         // global addr [31:0]
    g0.w = (unsigned)((ga >> 32) & 0x01FFFFFFull) | (2u << 30);  // [56:32]|type=2
    v8i_t g1;
    g1[0] = (int)((1u << 16)                     // data_size = 1 (2 bytes)
                | (1u << 20)                     // pad_enable
                | (pad_interval_code << 22)
                | (pad_amount_code << 25));
    g1[1] = (int)((tensor_d0 & 0xFFFFu) << 16);                  // dim0[15:0]
    g1[2] = (int)(((tensor_d0 >> 16) & 0xFFFFu)                  // dim0[31:16]
                | ((tensor_d1 & 0xFFFFu) << 16));                // dim1[15:0]
    g1[3] = (int)(((tensor_d1 >> 16) & 0xFFFFu)                  // dim1[31:16]
                | ((tile_d0 & 0xFFFFu) << 16));                  // tile_dim0
    g1[4] = (int)(tile_d1 & 0xFFFFu);                            // tile_dim1 (dim2=0)
    g1[5] = (int)(unsigned)(stride0 & 0xFFFFFFFFull);            // stride0[31:0]
    g1[6] = (int)(unsigned)((stride0 >> 32) & 0xFFFFull);        // stride0[47:32]
    g1[7] = 0;                                                   // stride1 = 0 (2D)
    v4i_t gz4 = {0, 0, 0, 0};
    v8i_t gz8 = {0, 0, 0, 0, 0, 0, 0, 0};
    __builtin_amdgcn_tensor_load_to_lds(g0, g1, gz4, gz4, gz8, 0);
}

// ---------------------------------------------------------------------------
// Tiled GEMM (bf16 x bf16 -> f32 accum):  C = (relu?)(A @ B + bias)
//  A: [M,K] bf16 row-major, B: [K,N] bf16 row-major, K%32==0, N%128==0.
//  Block 128x128, 256 threads = 8 waves (4x2), each wave 32x64 = 2x4 WMMA.
//  Tiles staged global->LDS by the Tensor Data Mover, double buffered;
//  wave 0 drives the TDM, TENSORcnt ordering gives the pipeline.
//  Output: f32 (Cf) and/or bf16 (Cb).
// ---------------------------------------------------------------------------
#define MBLK 128
#define NBLK 128
#define KSTEP 32
#define ASTRIDE 40    // 32 + 8 ushort  (TDM pad: 16 dwords -> +4 dwords)
#define BSTRIDE 136   // 128 + 8 ushort (TDM pad: 64 dwords -> +4 dwords)

__global__ __launch_bounds__(256)
void gemm_bf16_wmma_tdm(const unsigned short* __restrict__ A,
                        const unsigned short* __restrict__ B,
                        const float* __restrict__ bias,
                        float* __restrict__ Cf, unsigned short* __restrict__ Cb,
                        int M, int K, int N, int relu)
{
    __shared__ unsigned short As[2][MBLK][ASTRIDE];   // 2 x 10240 B
    __shared__ unsigned short Bs[2][KSTEP][BSTRIDE];  // 2 x  8704 B

    const int tid   = threadIdx.x;
    const int lane  = tid & 31;
    const int wid   = tid >> 5;
    const int wm    = wid & 3;
    const int wn    = wid >> 2;
    const int frow  = lane & 15;
    const int fhalf = lane >> 4;

    const int blockM = blockIdx.y * MBLK;
    const int blockN = blockIdx.x * NBLK;

    const unsigned aoff[2] = { (unsigned)(size_t)&As[0][0][0],
                               (unsigned)(size_t)&As[1][0][0] };
    const unsigned boff[2] = { (unsigned)(size_t)&Bs[0][0][0],
                               (unsigned)(size_t)&Bs[1][0][0] };

    v8f acc[2][4];
#pragma unroll
    for (int mi = 0; mi < 2; ++mi)
#pragma unroll
        for (int ni = 0; ni < 4; ++ni) acc[mi][ni] = (v8f)0.0f;

    const int KT = K / KSTEP;

    // prologue: stage k-step 0 into buffer 0
    if (wid == 0) {
        tdm_load_2d_bf16(aoff[0], A + (size_t)blockM * K,
                         (unsigned)K, (unsigned)(M - blockM),
                         KSTEP, MBLK, (unsigned long long)K, 3u, 3u);
        tdm_load_2d_bf16(boff[0], B + blockN,
                         (unsigned)(N - blockN), (unsigned)K,
                         NBLK, KSTEP, (unsigned long long)N, 5u, 3u);
    }

    for (int kt = 0; kt < KT; ++kt) {
        const int cur = kt & 1;
        if (wid == 0) {
            if (kt + 1 < KT) {
                const int k1 = (kt + 1) * KSTEP;
                tdm_load_2d_bf16(aoff[cur ^ 1],
                                 A + (size_t)blockM * K + k1,
                                 (unsigned)(K - k1), (unsigned)(M - blockM),
                                 KSTEP, MBLK, (unsigned long long)K, 3u, 3u);
                tdm_load_2d_bf16(boff[cur ^ 1],
                                 B + (size_t)k1 * N + blockN,
                                 (unsigned)(N - blockN), (unsigned)(K - k1),
                                 NBLK, KSTEP, (unsigned long long)N, 5u, 3u);
                __builtin_amdgcn_s_wait_tensorcnt(2);  // current pair landed
            } else {
                __builtin_amdgcn_s_wait_tensorcnt(0);
            }
        }
        __syncthreads();   // tile `cur` visible to all waves

        v16bf afrag[2];
#pragma unroll
        for (int mi = 0; mi < 2; ++mi) {
            const int row = wm * 32 + mi * 16 + frow;
            v8bf lo = *(const v8bf*)&As[cur][row][fhalf * 8];       // K h*8+0..7
            v8bf hi = *(const v8bf*)&As[cur][row][fhalf * 8 + 16];  // K h*8+16..23
            afrag[mi] = cat8(lo, hi);
        }
        v16bf bfrag[4];
#pragma unroll
        for (int ni = 0; ni < 4; ++ni) {
            const int col = wn * 64 + ni * 16 + frow;
#pragma unroll
            for (int s = 0; s < 16; ++s) {
                const int kk = (s < 8) ? (fhalf * 8 + s) : (16 + fhalf * 8 + (s - 8));
                bfrag[ni][s] = __builtin_bit_cast(__bf16, Bs[cur][kk][col]);
            }
        }

#pragma unroll
        for (int mi = 0; mi < 2; ++mi)
#pragma unroll
            for (int ni = 0; ni < 4; ++ni)
                acc[mi][ni] = __builtin_amdgcn_wmma_f32_16x16x32_bf16(
                    false, afrag[mi], false, bfrag[ni],
                    (short)0, acc[mi][ni], false, false);

        __syncthreads();   // done reading `cur` before TDM overwrites it
    }

    // epilogue: C layout M = v + 8*fhalf, N = frow
#pragma unroll
    for (int mi = 0; mi < 2; ++mi) {
#pragma unroll
        for (int ni = 0; ni < 4; ++ni) {
            const int col = blockN + wn * 64 + ni * 16 + frow;
            const float bv = bias ? bias[col] : 0.0f;
            const int rbase = blockM + wm * 32 + mi * 16 + fhalf * 8;
#pragma unroll
            for (int v = 0; v < 8; ++v) {
                const int grow = rbase + v;
                if (grow < M) {
                    float x = acc[mi][ni][v] + bv;
                    if (relu) x = fmaxf(x, 0.0f);
                    if (Cf) Cf[(size_t)grow * N + col] = x;
                    if (Cb) Cb[(size_t)grow * N + col] = f32_to_bf16_rne(x);
                }
            }
        }
    }
}

// ---------------------------------------------------------------------------
// Elementwise f32 -> bf16 (n multiple of 4)
// ---------------------------------------------------------------------------
__global__ __launch_bounds__(256)
void cvt_f32_bf16(const float* __restrict__ in, unsigned short* __restrict__ out,
                  int n)
{
    const int i = (blockIdx.x * blockDim.x + threadIdx.x) * 4;
    if (i >= n) return;
    const float4 f = *(const float4*)(in + i);
    ushort4 o;
    o.x = f32_to_bf16_rne(f.x);
    o.y = f32_to_bf16_rne(f.y);
    o.z = f32_to_bf16_rne(f.z);
    o.w = f32_to_bf16_rne(f.w);
    *(ushort4*)(out + i) = o;
}

// ---------------------------------------------------------------------------
// Graph helpers
// ---------------------------------------------------------------------------
__global__ void init_deg(float* deg, int n) {
    int i = blockIdx.x * blockDim.x + threadIdx.x;
    if (i < n) deg[i] = 1.0f;  // self-loop
}

__global__ void deg_accum(const int* __restrict__ dst, int e, float* deg) {
    int i = blockIdx.x * blockDim.x + threadIdx.x;
    if (i < e) atomicAdd(&deg[dst[i]], 1.0f);
}

__global__ void deg_to_dinv(float* deg, int n) {
    int i = blockIdx.x * blockDim.x + threadIdx.x;
    if (i < n) deg[i] = rsqrtf(deg[i]);
}

// agg[dst] += dinv[src]*dinv[dst] * h[src]   (F = 512, 4 floats / thread)
__global__ __launch_bounds__(256)
void gcn_scatter(const float* __restrict__ h, const int* __restrict__ src,
                 const int* __restrict__ dst, const float* __restrict__ dinv,
                 float* __restrict__ agg, int E)
{
    const unsigned t = blockIdx.x * 256u + threadIdx.x;
    const unsigned e = t >> 7;
    if (e >= (unsigned)E) return;
    const int f = (int)(t & 127u) * 4;
    const int s = src[e], d = dst[e];
    const float norm = dinv[s] * dinv[d];
    const float4 hv = *(const float4*)&h[(size_t)s * 512 + f];
    float* ap = &agg[(size_t)d * 512 + f];
    atomicAdd(ap + 0, norm * hv.x);
    atomicAdd(ap + 1, norm * hv.y);
    atomicAdd(ap + 2, norm * hv.z);
    atomicAdd(ap + 3, norm * hv.w);
}

// agg += dinv^2 * h + b ; optionally also emit bf16 copy for the next GEMM
__global__ __launch_bounds__(256)
void gcn_self_bias(const float* __restrict__ h, const float* __restrict__ dinv,
                   const float* __restrict__ b, float* __restrict__ agg,
                   unsigned short* __restrict__ agg_bf, int total)
{
    int i = blockIdx.x * blockDim.x + threadIdx.x;
    if (i >= total) return;
    const int row = i >> 9;
    const int col = i & 511;
    const float di = dinv[row];
    const float x = agg[i] + di * di * h[i] + b[col];
    agg[i] = x;
    if (agg_bf) agg_bf[i] = f32_to_bf16_rne(x);
}

// out[row] = sigmoid(cos(dp_row, tp_row)); one wave32 per row
__global__ __launch_bounds__(256)
void cosine_sigmoid(const float* __restrict__ dp, const float* __restrict__ tp,
                    float* __restrict__ out, int n)
{
    const int wid = threadIdx.x >> 5;
    const int lane = threadIdx.x & 31;
    const int row = blockIdx.x * 8 + wid;
    if (row >= n) return;
    const float* a = dp + (size_t)row * 512;
    const float* b = tp + (size_t)row * 512;
    float dot = 0.f, na = 0.f, nb = 0.f;
#pragma unroll 4
    for (int i = lane; i < 512; i += 32) {
        float x = a[i], y = b[i];
        dot += x * y; na += x * x; nb += y * y;
    }
#pragma unroll
    for (int off = 16; off > 0; off >>= 1) {
        dot += __shfl_down(dot, off, 32);
        na  += __shfl_down(na,  off, 32);
        nb  += __shfl_down(nb,  off, 32);
    }
    if (lane == 0) {
        float dn = fmaxf(sqrtf(na), 1e-8f);
        float tn = fmaxf(sqrtf(nb), 1e-8f);
        float c  = dot / (dn * tn);
        out[row] = 1.0f / (1.0f + __expf(-c));
    }
}

// ---------------------------------------------------------------------------
// Launch
// ---------------------------------------------------------------------------
extern "C" void kernel_launch(void* const* d_in, const int* in_sizes, int n_in,
                              void* d_out, int out_size, void* d_ws, size_t ws_size,
                              hipStream_t stream)
{
    const float* drug   = (const float*)d_in[0];   // [N,2048]
    const float* target = (const float*)d_in[1];   // [N,1024]
    const int*   dei    = (const int*)d_in[2];     // [2,E]
    const int*   tei    = (const int*)d_in[3];     // [2,E]
    const float* Wd     = (const float*)d_in[4];   // [2048,1024]
    const float* bd     = (const float*)d_in[5];
    const float* Wt     = (const float*)d_in[6];   // [1024,1024]
    const float* bt     = (const float*)d_in[7];
    const float* W1     = (const float*)d_in[8];   // [1024,512]
    const float* b1     = (const float*)d_in[9];
    const float* W2     = (const float*)d_in[10];  // [512,512]
    const float* b2     = (const float*)d_in[11];

    const int Nn = in_sizes[0] / 2048;
    const int E  = in_sizes[2] / 2;

    const int* src_d = dei;  const int* dst_d = dei + E;
    const int* src_t = tei;  const int* dst_t = tei + E;

    // ---- workspace carve-up (byte-based) ----
    char* p = (char*)d_ws;
    auto alloc = [&](size_t bytes) { char* r = p; p += (bytes + 255) & ~(size_t)255; return r; };

    unsigned short* drug_bf   = (unsigned short*)alloc((size_t)Nn * 2048 * 2);
    unsigned short* target_bf = (unsigned short*)alloc((size_t)Nn * 1024 * 2);
    unsigned short* dp_bf     = (unsigned short*)alloc((size_t)Nn * 1024 * 2); // also h2_d (f32)
    unsigned short* tp_bf     = (unsigned short*)alloc((size_t)Nn * 1024 * 2); // also h2_t (f32)
    float*          h_d       = (float*)alloc((size_t)Nn * 512 * 4);           // also agg2_d
    float*          h_t       = (float*)alloc((size_t)Nn * 512 * 4);           // also agg2_t
    float*          agg_d     = (float*)alloc((size_t)Nn * 512 * 4);
    float*          agg_t     = (float*)alloc((size_t)Nn * 512 * 4);
    unsigned short* aggbf_d   = (unsigned short*)alloc((size_t)Nn * 512 * 2);
    unsigned short* aggbf_t   = (unsigned short*)alloc((size_t)Nn * 512 * 2);
    unsigned short* Wd_bf     = (unsigned short*)alloc((size_t)2048 * 1024 * 2);
    unsigned short* Wt_bf     = (unsigned short*)alloc((size_t)1024 * 1024 * 2);
    unsigned short* W1_bf     = (unsigned short*)alloc((size_t)1024 * 512 * 2);
    unsigned short* W2_bf     = (unsigned short*)alloc((size_t)512 * 512 * 2);
    float*          dinv_d    = (float*)alloc((size_t)Nn * 4);
    float*          dinv_t    = (float*)alloc((size_t)Nn * 4);

    float* h2_d = (float*)dp_bf;   // N*1024*2B == N*512*4B, dp_bf dead by then
    float* h2_t = (float*)tp_bf;

    const int T = 256;
    const int nBlkNode = (Nn + T - 1) / T;
    const int nBlkEdge = (E + T - 1) / T;
    const unsigned nBlkScat = (unsigned)(((size_t)E * 128 + T - 1) / T);
    const int totElem = Nn * 512;
    const int nBlkElem = (totElem + T - 1) / T;
    const int gridMy = (Nn + MBLK - 1) / MBLK;
    auto cvtBlk = [&](size_t n) { return (unsigned)((n / 4 + T - 1) / T); };

    // 1) degree -> dinv for both graphs
    init_deg<<<nBlkNode, T, 0, stream>>>(dinv_d, Nn);
    init_deg<<<nBlkNode, T, 0, stream>>>(dinv_t, Nn);
    deg_accum<<<nBlkEdge, T, 0, stream>>>(dst_d, E, dinv_d);
    deg_accum<<<nBlkEdge, T, 0, stream>>>(dst_t, E, dinv_t);
    deg_to_dinv<<<nBlkNode, T, 0, stream>>>(dinv_d, Nn);
    deg_to_dinv<<<nBlkNode, T, 0, stream>>>(dinv_t, Nn);

    // 2) one-time f32 -> bf16 conversions (inputs + weights)
    cvt_f32_bf16<<<cvtBlk((size_t)Nn * 2048), T, 0, stream>>>(drug, drug_bf, Nn * 2048);
    cvt_f32_bf16<<<cvtBlk((size_t)Nn * 1024), T, 0, stream>>>(target, target_bf, Nn * 1024);
    cvt_f32_bf16<<<cvtBlk((size_t)2048 * 1024), T, 0, stream>>>(Wd, Wd_bf, 2048 * 1024);
    cvt_f32_bf16<<<cvtBlk((size_t)1024 * 1024), T, 0, stream>>>(Wt, Wt_bf, 1024 * 1024);
    cvt_f32_bf16<<<cvtBlk((size_t)1024 * 512), T, 0, stream>>>(W1, W1_bf, 1024 * 512);
    cvt_f32_bf16<<<cvtBlk((size_t)512 * 512), T, 0, stream>>>(W2, W2_bf, 512 * 512);

    // 3) projectors: Linear + ReLU  -> bf16 activations
    gemm_bf16_wmma_tdm<<<dim3(1024 / NBLK, gridMy), T, 0, stream>>>(
        drug_bf, Wd_bf, bd, nullptr, dp_bf, Nn, 2048, 1024, 1);
    gemm_bf16_wmma_tdm<<<dim3(1024 / NBLK, gridMy), T, 0, stream>>>(
        target_bf, Wt_bf, bt, nullptr, tp_bf, Nn, 1024, 1024, 1);

    // 4) GCN layer 1 (W1: 1024 -> 512)
    gemm_bf16_wmma_tdm<<<dim3(512 / NBLK, gridMy), T, 0, stream>>>(
        dp_bf, W1_bf, nullptr, h_d, nullptr, Nn, 1024, 512, 0);
    (void)hipMemsetAsync(agg_d, 0, (size_t)totElem * sizeof(float), stream);
    gcn_scatter<<<nBlkScat, T, 0, stream>>>(h_d, src_d, dst_d, dinv_d, agg_d, E);
    gcn_self_bias<<<nBlkElem, T, 0, stream>>>(h_d, dinv_d, b1, agg_d, aggbf_d, totElem);

    gemm_bf16_wmma_tdm<<<dim3(512 / NBLK, gridMy), T, 0, stream>>>(
        tp_bf, W1_bf, nullptr, h_t, nullptr, Nn, 1024, 512, 0);
    (void)hipMemsetAsync(agg_t, 0, (size_t)totElem * sizeof(float), stream);
    gcn_scatter<<<nBlkScat, T, 0, stream>>>(h_t, src_t, dst_t, dinv_t, agg_t, E);
    gcn_self_bias<<<nBlkElem, T, 0, stream>>>(h_t, dinv_t, b1, agg_t, aggbf_t, totElem);

    // 5) GCN layer 2 (W2: 512 -> 512); reuse dp_bf/tp_bf as h2 (f32), h_d/h_t as agg2
    gemm_bf16_wmma_tdm<<<dim3(512 / NBLK, gridMy), T, 0, stream>>>(
        aggbf_d, W2_bf, nullptr, h2_d, nullptr, Nn, 512, 512, 0);
    (void)hipMemsetAsync(h_d, 0, (size_t)totElem * sizeof(float), stream);
    gcn_scatter<<<nBlkScat, T, 0, stream>>>(h2_d, src_d, dst_d, dinv_d, h_d, E);
    gcn_self_bias<<<nBlkElem, T, 0, stream>>>(h2_d, dinv_d, b2, h_d, nullptr, totElem);

    gemm_bf16_wmma_tdm<<<dim3(512 / NBLK, gridMy), T, 0, stream>>>(
        aggbf_t, W2_bf, nullptr, h2_t, nullptr, Nn, 512, 512, 0);
    (void)hipMemsetAsync(h_t, 0, (size_t)totElem * sizeof(float), stream);
    gcn_scatter<<<nBlkScat, T, 0, stream>>>(h2_t, src_t, dst_t, dinv_t, h_t, E);
    gcn_self_bias<<<nBlkElem, T, 0, stream>>>(h2_t, dinv_t, b2, h_t, nullptr, totElem);

    // 6) cosine similarity + sigmoid
    cosine_sigmoid<<<(Nn + 7) / 8, T, 0, stream>>>(h_d, h_t, (float*)d_out, Nn);
}
